// BaseModel_8211977470114
// MI455X (gfx1250) — compile-verified, matrix operand order checked by reading, not run on hardware
//
#include <hip/hip_runtime.h>
#include <stdint.h>
#include <stddef.h>

#define BSZ   64
#define TLEN  700
#define MROWS (BSZ * TLEN)   // 44800

typedef __attribute__((ext_vector_type(16))) __bf16 v16bf;
typedef __attribute__((ext_vector_type(8)))  float  v8f;

union Frag {
    v16bf v;
    uint4 q[2];
    unsigned short u[16];
};

__device__ __forceinline__ unsigned short f2bf(float f) {
    union { float f; unsigned u; } x; x.f = f;
    unsigned r = x.u + 0x7FFFu + ((x.u >> 16) & 1u);
    return (unsigned short)(r >> 16);
}
__device__ __forceinline__ float bf2f(unsigned short h) {
    union { unsigned u; float f; } x; x.u = ((unsigned)h) << 16;
    return x.f;
}
__device__ __forceinline__ float sigmoidf_(float x) {
    return 1.0f / (1.0f + __expf(-x));
}
__device__ __forceinline__ float tanhf_(float x) {
    x = fminf(fmaxf(x, -15.0f), 15.0f);
    float e = __expf(2.0f * x);
    return (e - 1.0f) / (e + 1.0f);
}
// load 32 contiguous K-values of a row into the CDNA5 16x16x32 fragment layout:
// element e (0..7) -> K = kh*8+e ; element e (8..15) -> K = 16+kh*8+(e-8); kh = lane/16
__device__ __forceinline__ void load_frag(Frag& f, const unsigned short* rowk, int kh) {
    f.q[0] = *(const uint4*)(rowk + kh * 8);
    f.q[1] = *(const uint4*)(rowk + 16 + kh * 8);
}

// ---------------------------------------------------------------- utilities
__global__ void k_zero32(unsigned* p, int n) {
    int i = blockIdx.x * blockDim.x + threadIdx.x;
    if (i < n) p[i] = 0u;
}

// ---------------------------------------------------------------- embedding
__global__ void k_embed(const float* __restrict__ x, const float* __restrict__ emb,
                        float* __restrict__ x0) {
    int idx = blockIdx.x * blockDim.x + threadIdx.x;
    if (idx >= MROWS) return;
    int b = idx / TLEN, t = idx % TLEN;
    const float* xb = x + (size_t)b * 51 * TLEN + t;
    float best = xb[0]; int am = 0;
    for (int c = 1; c < 22; c++) {
        float v = xb[(size_t)c * TLEN];
        if (v > best) { best = v; am = c; }
    }
    float* ob = x0 + (size_t)b * 51 * TLEN + t;
    for (int c = 0; c < 22; c++) ob[(size_t)c * TLEN] = emb[am * 22 + c];
    for (int c = 22; c < 51; c++) ob[(size_t)c * TLEN] = xb[(size_t)c * TLEN];
}

// ---------------------------------------------------------------- front convs
__global__ void k_front(const float* __restrict__ x0,
                        const float* __restrict__ c3w, const float* __restrict__ c3b,
                        const float* __restrict__ c5w, const float* __restrict__ c5b,
                        unsigned short* __restrict__ xin,
                        unsigned short* __restrict__ xcat1,
                        unsigned short* __restrict__ xcat2) {
    int row = blockIdx.x;          // b*T + t
    int b = row / TLEN, t = row % TLEN;
    const float* xb = x0 + (size_t)b * 51 * TLEN;
    int col = threadIdx.x;         // 256 threads, one column each
    float v = 0.f;
    if (col < 51) {
        v = xb[(size_t)col * TLEN + t];
    } else if (col < 151) {
        int o = col - 51;
        float acc = c3b[o];
        const float* w = c3w + o * 51 * 3;
        for (int c = 0; c < 51; c++) {
            const float* xc = xb + (size_t)c * TLEN;
            float xm = (t >= 1) ? xc[t - 1] : 0.f;
            float xz = xc[t];
            float xp = (t + 1 < TLEN) ? xc[t + 1] : 0.f;
            acc += xm * w[c * 3 + 0] + xz * w[c * 3 + 1] + xp * w[c * 3 + 2];
        }
        v = acc;
    } else if (col < 251) {
        int o = col - 151;
        float acc = c5b[o];
        const float* w = c5w + o * 51 * 5;
        for (int c = 0; c < 51; c++) {
            const float* xc = xb + (size_t)c * TLEN;
#pragma unroll
            for (int kk = 0; kk < 5; kk++) {
                int tt = t + kk - 2;
                float xv = (tt >= 0 && tt < TLEN) ? xc[tt] : 0.f;
                acc += xv * w[c * 5 + kk];
            }
        }
        v = acc;
    }
    if (col < 251) v = fmaxf(v, 0.f);
    unsigned short h = f2bf(v);
    xin[(size_t)row * 256 + col] = (col < 251) ? h : (unsigned short)0;
    if (col < 251) xcat1[(size_t)row * 768 + col] = h;
    if (col < 17)  xcat1[(size_t)row * 768 + 751 + col] = 0;
    if (col < 36)  xcat2[(size_t)row * 1536 + 1500 + col] = 0;
}

// ---------------------------------------------------------------- GRU weight prep
// Combined weight Wc [N4 = 4*Hpad rows, Kc = Dpad+Hpad cols] row-major, bf16.
// column groups: g0 = r (Wi & Wh), g1 = z (Wi & Wh), g2 = n_x (Wi only), g3 = n_h (Wh only)
__global__ void k_prep_gru(const float* __restrict__ Wi, const float* __restrict__ Wh,
                           const float* __restrict__ bi, const float* __restrict__ bh,
                           int D, int H, int Dpad, int Hpad,
                           unsigned short* __restrict__ Wc, float* __restrict__ bias4) {
    int n = blockIdx.x;                 // 0 .. 4*Hpad-1
    int Kc = Dpad + Hpad;
    int g = n / Hpad, h = n % Hpad;
    unsigned short* dst = Wc + (size_t)n * Kc;
    for (int k = threadIdx.x; k < Kc; k += blockDim.x) {
        float v = 0.f;
        if (h < H) {
            if (g <= 1) {
                int r = g * H + h;
                if (k < D) v = Wi[(size_t)r * D + k];
                else if (k >= Dpad && k < Dpad + H) v = Wh[(size_t)r * H + (k - Dpad)];
            } else if (g == 2) {
                if (k < D) v = Wi[(size_t)(2 * H + h) * D + k];
            } else {
                if (k >= Dpad && k < Dpad + H) v = Wh[(size_t)(2 * H + h) * H + (k - Dpad)];
            }
        }
        dst[k] = f2bf(v);
    }
    if (threadIdx.x == 0) {
        float bv = 0.f;
        if (h < H) {
            if (g == 0)      bv = bi[h] + bh[h];
            else if (g == 1) bv = bi[H + h] + bh[H + h];
            else if (g == 2) bv = bi[2 * H + h];
            else             bv = bh[2 * H + h];
        }
        bias4[n] = bv;
    }
}

// ---------------------------------------------------------------- linear weight prep
__global__ void k_prep_lin(const float* __restrict__ W, const float* __restrict__ b,
                           int Nreal, int Kreal, int Kpad,
                           unsigned short* __restrict__ Wt, float* __restrict__ bt) {
    int n = blockIdx.x;                 // 0 .. Npad-1
    unsigned short* dst = Wt + (size_t)n * Kpad;
    for (int k = threadIdx.x; k < Kpad; k += blockDim.x) {
        float v = (n < Nreal && k < Kreal) ? W[(size_t)n * Kreal + k] : 0.f;
        dst[k] = f2bf(v);
    }
    if (threadIdx.x == 0) bt[n] = (n < Nreal) ? b[n] : 0.f;
}

// ---------------------------------------------------------------- generic WMMA GEMM
// Y[M, Npad] = A[M, K](bf16) * W[Npad, K]^T(bf16) + bias ; optional relu.
__device__ __forceinline__ void store_tile(v8f c, int mbase, int nbase, int lane,
                                           const float* bias, int relu,
                                           unsigned short* outB, int ldob,
                                           float* outF, int ldof, int NrealF) {
    int n  = nbase + (lane & 15);
    int m0 = mbase + 8 * (lane >> 4);
    float bv = bias[n];
#pragma unroll
    for (int v = 0; v < 8; v++) {
        float val = c[v] + bv;
        if (relu) val = fmaxf(val, 0.f);
        int m = m0 + v;
        if (outB) outB[(size_t)m * ldob + n] = f2bf(val);
        if (outF && n < NrealF) outF[(size_t)m * ldof + n] = val;
    }
}

__global__ void __launch_bounds__(256) k_gemm(const unsigned short* __restrict__ A, int lda,
                                              const unsigned short* __restrict__ W, int ldw,
                                              const float* __restrict__ bias, int K,
                                              unsigned short* outB, int ldob,
                                              float* outF, int ldof, int NrealF, int relu) {
    int lane = threadIdx.x & 31, w = threadIdx.x >> 5;
    int wm = w >> 1, wn = w & 1;
    int m0 = blockIdx.x * 128 + wm * 32;
    int n0 = blockIdx.y * 64 + wn * 32;
    int kh = lane >> 4, r = lane & 15;
    const unsigned short* arow0 = A + (size_t)(m0 + r) * lda;
    const unsigned short* arow1 = A + (size_t)(m0 + 16 + r) * lda;
    const unsigned short* brow0 = W + (size_t)(n0 + r) * ldw;
    const unsigned short* brow1 = W + (size_t)(n0 + 16 + r) * ldw;
    v8f c00 = {}, c01 = {}, c10 = {}, c11 = {};
    for (int k0 = 0; k0 < K; k0 += 32) {
        Frag a0, a1, b0, b1;
        load_frag(a0, arow0 + k0, kh);
        load_frag(a1, arow1 + k0, kh);
        load_frag(b0, brow0 + k0, kh);
        load_frag(b1, brow1 + k0, kh);
        c00 = __builtin_amdgcn_wmma_f32_16x16x32_bf16(false, a0.v, false, b0.v, (short)0, c00, false, false);
        c01 = __builtin_amdgcn_wmma_f32_16x16x32_bf16(false, a0.v, false, b1.v, (short)0, c01, false, false);
        c10 = __builtin_amdgcn_wmma_f32_16x16x32_bf16(false, a1.v, false, b0.v, (short)0, c10, false, false);
        c11 = __builtin_amdgcn_wmma_f32_16x16x32_bf16(false, a1.v, false, b1.v, (short)0, c11, false, false);
    }
    store_tile(c00, m0,      n0,      lane, bias, relu, outB, ldob, outF, ldof, NrealF);
    store_tile(c01, m0,      n0 + 16, lane, bias, relu, outB, ldob, outF, ldof, NrealF);
    store_tile(c10, m0 + 16, n0,      lane, bias, relu, outB, ldob, outF, ldof, NrealF);
    store_tile(c11, m0 + 16, n0 + 16, lane, bias, relu, outB, ldob, outF, ldof, NrealF);
}

// ---------------------------------------------------------------- persistent GRU scan
// grid = 2*blocksPerDir blocks; block = 256 thr (8 waves).
// Each block owns h-slice [h0, h0+32) of one direction. Its combined-weight slice
// (128 rows x Kc) is staged ONCE into LDS (row stride padded +8 elems for banking)
// and re-read with ds_load_b128 for all 700 steps; gates = [x_t, h_{t-1}] @ W^T via
// WMMA; pointwise GRU update; device-wide phase barrier per step.
__global__ void __launch_bounds__(256) k_gru_scan(
        const unsigned short* __restrict__ X, int Dpad,
        const unsigned short* __restrict__ Wc, const float* __restrict__ bias4,
        int Hpad, int Kc,
        unsigned short* __restrict__ hbuf,
        unsigned short* out1, int ldo1, int co1f, int co1b, int flip1, int hw1, int dm1,
        unsigned short* out2, int ldo2, int co2f, int co2b, int flip2, int hw2, int dm2,
        unsigned* __restrict__ cnt, int blocksPerDir) {
    extern __shared__ unsigned char smem[];
    int KcL = Kc + 8;                                   // padded LDS row stride (elems)
    unsigned short* wlds = (unsigned short*)smem;       // [128][KcL] bf16 weights
    float* gls = (float*)(smem + (size_t)128 * KcL * 2);// [64][128] gate pre-acts

    int dir = blockIdx.x / blocksPerDir;
    int blk = blockIdx.x % blocksPerDir;
    int h0 = blk * 32;
    int nb = gridDim.x;
    const unsigned short* Wd = Wc + (size_t)dir * 4 * Hpad * Kc;
    const float* bd = bias4 + (size_t)dir * 4 * Hpad;
    unsigned short* hd = hbuf + (size_t)dir * 2 * 64 * Hpad;

    // ---- stage this block's weight slice into LDS (rows: g*32+hh -> n = g*Hpad+h0+hh)
    {
        int q8 = Kc / 8;                    // uint4 chunks per row
        int total = 128 * q8;
        for (int idx = threadIdx.x; idx < total; idx += 256) {
            int ln = idx / q8, c8 = idx % q8;
            int ngl = (ln >> 5) * Hpad + h0 + (ln & 31);
            uint4 val = *(const uint4*)(Wd + (size_t)ngl * Kc + c8 * 8);
            *(uint4*)(wlds + (size_t)ln * KcL + c8 * 8) = val;
        }
    }
    __syncthreads();

    int lane = threadIdx.x & 31, w = threadIdx.x >> 5;
    int mtile = w >> 1, nhalf = w & 1;
    int kh = lane >> 4, r = lane & 15;
    int b = mtile * 16 + r;                 // batch row this lane loads for A

    const unsigned short* wrow[4];
#pragma unroll
    for (int j = 0; j < 4; j++) {
        int nt = nhalf * 4 + j;             // block-local N-tile 0..7
        wrow[j] = wlds + (size_t)(nt * 16 + r) * KcL;
    }
    int coff1 = dir ? co1b : co1f;
    int coff2 = dir ? co2b : co2f;
    v8f zacc = {};

    for (int s = 0; s < TLEN; s++) {
        int t_in = dir ? (TLEN - 1 - s) : s;
        const unsigned short* xrow = X + ((size_t)b * TLEN + t_in) * Dpad;
        const unsigned short* hprev = hd + (size_t)(s & 1) * 64 * Hpad;
        unsigned short* hnew = hd + (size_t)((s + 1) & 1) * 64 * Hpad;

        v8f acc[4];
#pragma unroll
        for (int j = 0; j < 4; j++) acc[j] = zacc;

        for (int k0 = 0; k0 < Kc; k0 += 32) {
            Frag a;
            if (k0 < Dpad) {
                load_frag(a, xrow + k0, kh);
            } else if (s == 0) {
                a.q[0] = make_uint4(0, 0, 0, 0);
                a.q[1] = make_uint4(0, 0, 0, 0);
            } else {
                load_frag(a, hprev + (size_t)b * Hpad + (k0 - Dpad), kh);
            }
#pragma unroll
            for (int j = 0; j < 4; j++) {
                Frag bb;
                load_frag(bb, wrow[j] + k0, kh);   // LDS reads (ds_load_b128)
                acc[j] = __builtin_amdgcn_wmma_f32_16x16x32_bf16(false, a.v, false, bb.v,
                                                                 (short)0, acc[j], false, false);
            }
        }
        // spill gate pre-activations to LDS: gls[m][nt*16 + r]
#pragma unroll
        for (int j = 0; j < 4; j++) {
            int col = (nhalf * 4 + j) * 16 + r;
            int mb = mtile * 16 + 8 * kh;
#pragma unroll
            for (int v = 0; v < 8; v++) gls[(size_t)(mb + v) * 128 + col] = acc[j][v];
        }
        __syncthreads();

        // pointwise GRU update: 64 x 32 elements, 8 per thread
#pragma unroll
        for (int i = 0; i < 8; i++) {
            int idx = threadIdx.x + i * 256;
            int m = idx >> 5, hc = idx & 31;
            int hg = h0 + hc;
            float rr = sigmoidf_(gls[(size_t)m * 128 + hc]      + bd[hg]);
            float zz = sigmoidf_(gls[(size_t)m * 128 + 32 + hc] + bd[Hpad + hg]);
            float nx = gls[(size_t)m * 128 + 64 + hc] + bd[2 * Hpad + hg];
            float nh = gls[(size_t)m * 128 + 96 + hc] + bd[3 * Hpad + hg];
            float hp = (s == 0) ? 0.f : bf2f(hprev[(size_t)m * Hpad + hg]);
            float nn = tanhf_(nx + rr * nh);
            float hv = (1.f - zz) * nn + zz * hp;
            unsigned short hb = f2bf(hv);
            hnew[(size_t)m * Hpad + hg] = hb;
            if (out1 && ((dm1 >> dir) & 1) && hg < hw1) {
                int t_out = (dir && flip1) ? (TLEN - 1 - s) : s;
                out1[((size_t)m * TLEN + t_out) * ldo1 + coff1 + hg] = hb;
            }
            if (out2 && ((dm2 >> dir) & 1) && hg < hw2) {
                int t_out = (dir && flip2) ? (TLEN - 1 - s) : s;
                out2[((size_t)m * TLEN + t_out) * ldo2 + coff2 + hg] = hb;
            }
        }
        // device-wide phase barrier (release h_t, acquire for next step)
        __threadfence();
        __syncthreads();
        if (threadIdx.x == 0) {
            atomicAdd(cnt, 1u);
            unsigned target = (unsigned)(s + 1) * (unsigned)nb;
            while (__hip_atomic_load(cnt, __ATOMIC_RELAXED, __HIP_MEMORY_SCOPE_AGENT) < target)
                __builtin_amdgcn_s_sleep(2);
        }
        __syncthreads();
        __threadfence();
    }
}

// ---------------------------------------------------------------- elementwise sum
__global__ void k_sum(const unsigned short* __restrict__ a, const unsigned short* __restrict__ b,
                      unsigned short* __restrict__ o, int n) {
    int i = blockIdx.x * blockDim.x + threadIdx.x;
    if (i < n) o[i] = f2bf(bf2f(a[i]) + bf2f(b[i]));
}

// ================================================================ host
extern "C" void kernel_launch(void* const* d_in, const int* in_sizes, int n_in,
                              void* d_out, int out_size, void* d_ws, size_t ws_size,
                              hipStream_t stream) {
    const float* x     = (const float*)d_in[0];
    const float* emb   = (const float*)d_in[1];
    const float* c3w   = (const float*)d_in[2];
    const float* c3b   = (const float*)d_in[3];
    const float* c5w   = (const float*)d_in[4];
    const float* c5b   = (const float*)d_in[5];
    const float* c11w  = (const float*)d_in[6];
    const float* c11b  = (const float*)d_in[7];
    const float* c12w  = (const float*)d_in[8];
    const float* c12b  = (const float*)d_in[9];
    const float* g1fW[4] = {(const float*)d_in[10], (const float*)d_in[11], (const float*)d_in[12], (const float*)d_in[13]};
    const float* g1bW[4] = {(const float*)d_in[14], (const float*)d_in[15], (const float*)d_in[16], (const float*)d_in[17]};
    const float* g2fW[4] = {(const float*)d_in[18], (const float*)d_in[19], (const float*)d_in[20], (const float*)d_in[21]};
    const float* g2bW[4] = {(const float*)d_in[22], (const float*)d_in[23], (const float*)d_in[24], (const float*)d_in[25]};
    const float* g3fW[4] = {(const float*)d_in[26], (const float*)d_in[27], (const float*)d_in[28], (const float*)d_in[29]};
    const float* g3bW[4] = {(const float*)d_in[30], (const float*)d_in[31], (const float*)d_in[32], (const float*)d_in[33]};
    const float* fc1w  = (const float*)d_in[34];
    const float* fc1b  = (const float*)d_in[35];
    const float* fc2w  = (const float*)d_in[36];
    const float* fc2b  = (const float*)d_in[37];
    (void)in_sizes; (void)n_in; (void)out_size; (void)ws_size;

    // -------- workspace layout
    char* base = (char*)d_ws;
    size_t off = 0;
    auto alloc = [&](size_t bytes) -> char* {
        char* p = base + off;
        off += (bytes + 255) & ~(size_t)255;
        return p;
    };
    float*          x0    = (float*)         alloc((size_t)BSZ * 51 * TLEN * 4);
    unsigned short* xin   = (unsigned short*)alloc((size_t)MROWS * 256 * 2);
    unsigned short* xcat1 = (unsigned short*)alloc((size_t)MROWS * 768 * 2);
    unsigned short* yc11  = (unsigned short*)alloc((size_t)MROWS * 512 * 2);
    unsigned short* xcat2 = (unsigned short*)alloc((size_t)MROWS * 1536 * 2);
    unsigned short* yc12  = (unsigned short*)alloc((size_t)MROWS * 512 * 2);
    unsigned short* h3f   = (unsigned short*)alloc((size_t)MROWS * 512 * 2);
    unsigned short* h3b   = (unsigned short*)alloc((size_t)MROWS * 512 * 2);
    unsigned short* fcin  = (unsigned short*)alloc((size_t)MROWS * 512 * 2);
    unsigned short* yfc1  = (unsigned short*)alloc((size_t)MROWS * 1024 * 2);
    unsigned short* Wg1   = (unsigned short*)alloc((size_t)2 * 1024 * 512 * 2);
    float*          Bg1   = (float*)         alloc((size_t)2 * 1024 * 4);
    unsigned short* Wg2   = (unsigned short*)alloc((size_t)2 * 2048 * 1024 * 2);
    float*          Bg2   = (float*)         alloc((size_t)2 * 2048 * 4);
    unsigned short* Wg3   = (unsigned short*)alloc((size_t)2 * 2048 * 1024 * 2);
    float*          Bg3   = (float*)         alloc((size_t)2 * 2048 * 4);
    unsigned short* Wc11  = (unsigned short*)alloc((size_t)512 * 768 * 2);
    float*          Bc11  = (float*)         alloc((size_t)512 * 4);
    unsigned short* Wc12  = (unsigned short*)alloc((size_t)512 * 1536 * 2);
    float*          Bc12  = (float*)         alloc((size_t)512 * 4);
    unsigned short* Wfc1  = (unsigned short*)alloc((size_t)1024 * 512 * 2);
    float*          Bfc1  = (float*)         alloc((size_t)1024 * 4);
    unsigned short* Wfc2  = (unsigned short*)alloc((size_t)64 * 1024 * 2);
    float*          Bfc2  = (float*)         alloc((size_t)64 * 4);
    unsigned short* hbuf  = (unsigned short*)alloc((size_t)2 * 2 * 64 * 512 * 2);
    unsigned*       cnts  = (unsigned*)      alloc(64 * 4);

    // -------- barrier counters
    k_zero32<<<1, 64, 0, stream>>>(cnts, 64);

    // -------- weight prep (bf16, padded, combined GRU layout)
    k_prep_gru<<<4 * 256, 128, 0, stream>>>(g1fW[0], g1fW[1], g1fW[2], g1fW[3],
                                            251, 250, 256, 256, Wg1, Bg1);
    k_prep_gru<<<4 * 256, 128, 0, stream>>>(g1bW[0], g1bW[1], g1bW[2], g1bW[3],
                                            251, 250, 256, 256, Wg1 + (size_t)1024 * 512, Bg1 + 1024);
    k_prep_gru<<<4 * 512, 128, 0, stream>>>(g2fW[0], g2fW[1], g2fW[2], g2fW[3],
                                            500, 500, 512, 512, Wg2, Bg2);
    k_prep_gru<<<4 * 512, 128, 0, stream>>>(g2bW[0], g2bW[1], g2bW[2], g2bW[3],
                                            500, 500, 512, 512, Wg2 + (size_t)2048 * 1024, Bg2 + 2048);
    k_prep_gru<<<4 * 512, 128, 0, stream>>>(g3fW[0], g3fW[1], g3fW[2], g3fW[3],
                                            500, 500, 512, 512, Wg3, Bg3);
    k_prep_gru<<<4 * 512, 128, 0, stream>>>(g3bW[0], g3bW[1], g3bW[2], g3bW[3],
                                            500, 500, 512, 512, Wg3 + (size_t)2048 * 1024, Bg3 + 2048);
    k_prep_lin<<<512, 128, 0, stream>>>(c11w, c11b, 500, 751, 768, Wc11, Bc11);
    k_prep_lin<<<512, 128, 0, stream>>>(c12w, c12b, 500, 1500, 1536, Wc12, Bc12);
    k_prep_lin<<<1024, 128, 0, stream>>>(fc1w, fc1b, 1024, 500, 512, Wfc1, Bfc1);
    k_prep_lin<<<64, 128, 0, stream>>>(fc2w, fc2b, 9, 1024, 1024, Wfc2, Bfc2);

    // -------- front end
    k_embed<<<(MROWS + 255) / 256, 256, 0, stream>>>(x, emb, x0);
    k_front<<<MROWS, 256, 0, stream>>>(x0, c3w, c3b, c5w, c5b, xin, xcat1, xcat2);

    // dynamic LDS sizes: weights (128 rows x (Kc+8) bf16) + gates (64 x 128 f32)
    size_t shm1 = (size_t)128 * (512 + 8) * 2  + (size_t)64 * 128 * 4;   // 165888
    size_t shm2 = (size_t)128 * (1024 + 8) * 2 + (size_t)64 * 128 * 4;   // 296960

    // -------- BiGRU 1: in xin [M,256] -> O1 into xcat1[:,251:751] and xcat2[:,0:500]
    k_gru_scan<<<16, 256, shm1, stream>>>(xin, 256, Wg1, Bg1, 256, 512, hbuf,
                                          xcat1, 768, 251, 501, 1, 250, 3,
                                          xcat2, 1536, 0, 250, 1, 250, 3,
                                          cnts + 0, 8);
    // -------- conv1x1 #1: [M,768] x [512,768]^T -> yc11 (relu)
    k_gemm<<<dim3(MROWS / 128, 8), 256, 0, stream>>>(xcat1, 768, Wc11, 768, Bc11, 768,
                                                     yc11, 512, nullptr, 0, 0, 1);
    // -------- BiGRU 2: in yc11 [M,512] -> O2 into xcat2[:,500:1500]
    k_gru_scan<<<32, 256, shm2, stream>>>(yc11, 512, Wg2, Bg2, 512, 1024, hbuf,
                                          xcat2, 1536, 500, 1000, 1, 500, 3,
                                          nullptr, 0, 0, 0, 0, 0, 0,
                                          cnts + 1, 16);
    // -------- conv1x1 #2: [M,1536] x [512,1536]^T -> yc12 (relu)
    k_gemm<<<dim3(MROWS / 128, 8), 256, 0, stream>>>(xcat2, 1536, Wc12, 1536, Bc12, 1536,
                                                     yc12, 512, nullptr, 0, 0, 1);
    // -------- GRU 3 fwd + bwd (outputs in scan order, not re-flipped)
    k_gru_scan<<<32, 256, shm2, stream>>>(yc12, 512, Wg3, Bg3, 512, 1024, hbuf,
                                          h3f, 512, 0, 0, 0, 512, 1,
                                          h3b, 512, 0, 0, 0, 512, 2,
                                          cnts + 2, 16);
    // -------- sum
    k_sum<<<(MROWS * 512 + 255) / 256, 256, 0, stream>>>(h3f, h3b, fcin, MROWS * 512);
    // -------- FC1 (relu) and FC2 (f32 out, first 9 cols)
    k_gemm<<<dim3(MROWS / 128, 16), 256, 0, stream>>>(fcin, 512, Wfc1, 512, Bfc1, 512,
                                                      yfc1, 1024, nullptr, 0, 0, 1);
    k_gemm<<<dim3(MROWS / 128, 1), 256, 0, stream>>>(yfc1, 1024, Wfc2, 1024, Bfc2, 1024,
                                                     nullptr, 0, (float*)d_out, 9, 9, 0);
}
